// TMDLayer_1906965479482
// MI455X (gfx1250) — compile-verified, matrix operand order checked by reading, not run on hardware
//
#include <hip/hip_runtime.h>
#include <hip/hip_bf16.h>

// ---------------------------------------------------------------------------
// TMDLayer: out = (1-dt)*x + (dt/EPS) * diag(1/D) * ( K @ (d_vec * xf) )
//   K   = exp(-pairwise_sqdist(z)),  z = xf@proj_w+b   (4*EPS == 1)
//   d_vec = pi/q, q = rowsum(K), D = rowsum(K*d_vec)+1e-5
// Dominant cost: 4096^3 GEMM -> bf16 v_wmma_f32_16x16x32_bf16,
// double-buffered LDS staging, v_perm_b32 packed f32->bf16 truncation.
// ---------------------------------------------------------------------------

typedef __attribute__((ext_vector_type(8)))  float        v8f;
typedef __attribute__((ext_vector_type(8)))  unsigned int v8u;
typedef __attribute__((ext_vector_type(16))) __bf16       v16bf;

#define NN   4096   // n rows
#define FF   4096   // in_features (c*h*w)
#define LL   16     // latent dim

__device__ __forceinline__ unsigned short f2bf(float f) {
    unsigned int u = __builtin_bit_cast(unsigned int, f);
    u += 0x7FFFu + ((u >> 16) & 1u);           // round-to-nearest-even
    return (unsigned short)(u >> 16);
}
__device__ __forceinline__ float bf2f(unsigned short h) {
    return __builtin_bit_cast(float, (unsigned int)h << 16);
}
// RNE pack (cold paths).
__device__ __forceinline__ unsigned int pack2_bf16(float a, float b) {
    return (unsigned int)f2bf(a) | ((unsigned int)f2bf(b) << 16);
}
// Truncating pack: one v_perm_b32 grabs the high 16 bits of both floats.
// Result: lo16 = bf16(a), hi16 = bf16(b).
__device__ __forceinline__ unsigned int pack2_bf16_trunc(float a, float b) {
    unsigned int u0 = __builtin_bit_cast(unsigned int, a);
    unsigned int u1 = __builtin_bit_cast(unsigned int, b);
#if __has_builtin(__builtin_amdgcn_perm)
    return __builtin_amdgcn_perm(u1, u0, 0x07060302u);  // {u1.b3,u1.b2,u0.b3,u0.b2}
#else
    return (u0 >> 16) | (u1 & 0xFFFF0000u);
#endif
}

// ------------------------- stage 1: z = xf @ proj_w + b, sq = ||z||^2 ------
__global__ __launch_bounds__(256) void proj_kernel(
    const float* __restrict__ xf, const float* __restrict__ pw,
    const float* __restrict__ pb, float* __restrict__ z, float* __restrict__ sq)
{
    __shared__ float red[256 * LL];
    __shared__ float zr[LL];
    const int i = blockIdx.x, t = threadIdx.x;
    float acc[LL];
#pragma unroll
    for (int l = 0; l < LL; ++l) acc[l] = 0.f;
    for (int k = t; k < FF; k += 256) {
        const float xv = xf[(size_t)i * FF + k];
        const float* w = pw + (size_t)k * LL;
#pragma unroll
        for (int l = 0; l < LL; ++l) acc[l] += xv * w[l];
    }
#pragma unroll
    for (int l = 0; l < LL; ++l) red[t * LL + l] = acc[l];
    __syncthreads();
    if (t < LL) {
        float s = 0.f;
        for (int r = 0; r < 256; ++r) s += red[r * LL + t];
        s += pb[t];
        z[i * LL + t] = s;
        zr[t] = s * s;
    }
    __syncthreads();
    if (t == 0) {
        float s = 0.f;
#pragma unroll
        for (int l = 0; l < LL; ++l) s += zr[l];
        sq[i] = s;
    }
}

// ------------------- stage 2: K row (bf16) + q = rowsum(K) -----------------
__global__ __launch_bounds__(256) void kq_kernel(
    const float* __restrict__ z, const float* __restrict__ sq,
    unsigned short* __restrict__ Kbf, float* __restrict__ q)
{
    __shared__ float zi[LL];
    __shared__ float red[256];
    const int i = blockIdx.x, t = threadIdx.x;
    if (t < LL) zi[t] = z[i * LL + t];
    __syncthreads();
    const float sqi = sq[i];
    float qp = 0.f;
    for (int j = 2 * t; j < NN; j += 512) {          // 2 cols/thread -> packed store
        const float* zj = z + j * LL;
        float dot0 = 0.f, dot1 = 0.f;
#pragma unroll
        for (int l = 0; l < LL; ++l) { dot0 += zi[l] * zj[l]; dot1 += zi[l] * zj[LL + l]; }
        const float k0 = __expf(-(sqi + sq[j]     - 2.f * dot0));  // 4*EPS==1
        const float k1 = __expf(-(sqi + sq[j + 1] - 2.f * dot1));
        *(unsigned int*)&Kbf[(size_t)i * NN + j] = pack2_bf16(k0, k1);
        qp += k0 + k1;
    }
    red[t] = qp;
    __syncthreads();
    for (int s = 128; s > 0; s >>= 1) { if (t < s) red[t] += red[t + s]; __syncthreads(); }
    if (t == 0) q[i] = red[0];
}

// ---------- stage 3: pi = sigmoid(relu(z@W1+b1)@W2+b2); dvec = pi/q --------
__global__ __launch_bounds__(256) void pi_kernel(
    const float* __restrict__ z,  const float* __restrict__ w1,
    const float* __restrict__ b1, const float* __restrict__ w2,
    const float* __restrict__ b2, const float* __restrict__ q,
    float* __restrict__ dvec)
{
    __shared__ float zi[LL];
    __shared__ float red[256];
    const int i = blockIdx.x, t = threadIdx.x;
    if (t < LL) zi[t] = z[i * LL + t];
    __syncthreads();
    float p = 0.f;
    for (int j = t; j < FF; j += 256) {
        float s = b1[j];
#pragma unroll
        for (int l = 0; l < LL; ++l) s += zi[l] * w1[(size_t)l * FF + j];
        s = fmaxf(s, 0.f);
        p += s * w2[j];
    }
    red[t] = p;
    __syncthreads();
    for (int s = 128; s > 0; s >>= 1) { if (t < s) red[t] += red[t + s]; __syncthreads(); }
    if (t == 0) {
        const float tt = red[0] + b2[0];
        const float pi = 1.f / (1.f + __expf(-tt));
        dvec[i] = pi / q[i];
    }
}

// ------------------- stage 4: D = rowsum(K * dvec) + 1e-5 ------------------
__global__ __launch_bounds__(256) void dsum_kernel(
    const unsigned short* __restrict__ Kbf, const float* __restrict__ dvec,
    float* __restrict__ Dd)
{
    __shared__ float red[256];
    const int i = blockIdx.x, t = threadIdx.x;
    float p = 0.f;
    for (int j = t; j < NN; j += 256)
        p += bf2f(Kbf[(size_t)i * NN + j]) * dvec[j];
    red[t] = p;
    __syncthreads();
    for (int s = 128; s > 0; s >>= 1) { if (t < s) red[t] += red[t + s]; __syncthreads(); }
    if (t == 0) Dd[i] = red[0] + 1e-5f;
}

// -------- stage 5: out = (1-dt)*xf + (4*dt/D[i]) * (K @ (dvec*xf)) ---------
// Block tile 128x128, 8 waves, wave tile 32x64 (2x4 wmma 16x16x32 tiles).
// Double-buffered LDS: loads for tile k+1 issue before the WMMAs of tile k.
__global__ __launch_bounds__(256) void gemm_kernel(
    const unsigned short* __restrict__ Kbf, const float* __restrict__ xf,
    const float* __restrict__ dvec, const float* __restrict__ Dd,
    const float* __restrict__ dt, float* __restrict__ out)
{
    __shared__ unsigned short Asub[2][128 * 34];   // [m][k], padded stride 34
    __shared__ unsigned short Bsub[2][128 * 34];   // [n][k], padded stride 34

    const int tid  = threadIdx.x;
    const int wave = tid >> 5;
    const int lane = tid & 31;
    const int half = lane >> 4;       // lane group 0-15 / 16-31
    const int lr   = lane & 15;
    const int wm   = wave & 3;        // M sub-block (x32)
    const int wn   = wave >> 2;       // N sub-block (x64)
    const int bm   = blockIdx.y * 128;
    const int bn   = blockIdx.x * 128;

    v8f acc[2][4];
#pragma unroll
    for (int mi = 0; mi < 2; ++mi)
#pragma unroll
        for (int nj = 0; nj < 4; ++nj) acc[mi][nj] = (v8f)0.f;

    // staging assignments
    const int ar = tid >> 1;          // A: row 0..127
    const int ac = (tid & 1) * 16;    // A: k offset 0/16
    const int br = tid >> 3;          // B: k-row 0..31
    const int bc = (tid & 7) * 16;    // B: n base 0..112

    const unsigned short* gA0 = Kbf + (size_t)(bm + ar) * NN + ac;
    const float*          gB0 = xf + (size_t)br * FF + bn + bc;

    // ---- prologue: stage tile 0 into buffer 0 ----
    {
        uint4 a0 = *(const uint4*)(gA0);
        uint4 a1 = *(const uint4*)(gA0 + 8);
        unsigned int* dA = (unsigned int*)&Asub[0][ar * 34 + ac];
        dA[0] = a0.x; dA[1] = a0.y; dA[2] = a0.z; dA[3] = a0.w;
        dA[4] = a1.x; dA[5] = a1.y; dA[6] = a1.z; dA[7] = a1.w;
        const float dv = dvec[br];
#pragma unroll
        for (int p = 0; p < 4; ++p) {
            float4 v = *(const float4*)(gB0 + p * 4);
            unsigned int p01 = pack2_bf16_trunc(v.x * dv, v.y * dv);
            unsigned int p23 = pack2_bf16_trunc(v.z * dv, v.w * dv);
            Bsub[0][(bc + p * 4 + 0) * 34 + br] = (unsigned short)p01;
            Bsub[0][(bc + p * 4 + 1) * 34 + br] = (unsigned short)(p01 >> 16);
            Bsub[0][(bc + p * 4 + 2) * 34 + br] = (unsigned short)p23;
            Bsub[0][(bc + p * 4 + 3) * 34 + br] = (unsigned short)(p23 >> 16);
        }
    }

    int buf = 0;
    for (int k0 = 0; k0 < NN; k0 += 32) {
        const bool has_next = (k0 + 32) < NN;
        uint4 a0{}, a1{};
        float4 b0{}, b1{}, b2{}, b3{};
        float dv = 0.f;
        if (has_next) {   // issue next-tile global loads before the barrier
            const unsigned short* gA = gA0 + k0 + 32;
            a0 = *(const uint4*)(gA);
            a1 = *(const uint4*)(gA + 8);
            const float* gB = gB0 + (size_t)(k0 + 32) * FF;
            dv = dvec[k0 + 32 + br];
            b0 = *(const float4*)(gB);
            b1 = *(const float4*)(gB + 4);
            b2 = *(const float4*)(gB + 8);
            b3 = *(const float4*)(gB + 12);
            if (k0 + 64 < NN) __builtin_prefetch(gA + 32, 0, 0);  // global_prefetch_b8
        }
        __syncthreads();   // staged buffer `buf` ready for all waves

        // ---- per-lane fragments (16-bit A/B VGPR layout, ISA 7.12.2) ----
        v16bf afrag[2], bfrag[4];
#pragma unroll
        for (int mi = 0; mi < 2; ++mi) {
            const int row = wm * 32 + mi * 16 + lr;
            v8u u;
#pragma unroll
            for (int d = 0; d < 4; ++d) {
                u[d]     = *(const unsigned int*)&Asub[buf][row * 34 + half * 8 + 2 * d];
                u[d + 4] = *(const unsigned int*)&Asub[buf][row * 34 + 16 + half * 8 + 2 * d];
            }
            afrag[mi] = __builtin_bit_cast(v16bf, u);
        }
#pragma unroll
        for (int nj = 0; nj < 4; ++nj) {
            const int col = wn * 64 + nj * 16 + lr;
            v8u u;
#pragma unroll
            for (int d = 0; d < 4; ++d) {
                u[d]     = *(const unsigned int*)&Bsub[buf][col * 34 + half * 8 + 2 * d];
                u[d + 4] = *(const unsigned int*)&Bsub[buf][col * 34 + 16 + half * 8 + 2 * d];
            }
            bfrag[nj] = __builtin_bit_cast(v16bf, u);
        }
#pragma unroll
        for (int mi = 0; mi < 2; ++mi)
#pragma unroll
            for (int nj = 0; nj < 4; ++nj)
                acc[mi][nj] = __builtin_amdgcn_wmma_f32_16x16x32_bf16(
                    false, afrag[mi], false, bfrag[nj],
                    (short)0, acc[mi][nj], false, false);

        if (has_next) {    // store next tile into the other buffer (no race: barrier above)
            const int nb = buf ^ 1;
            unsigned int* dA = (unsigned int*)&Asub[nb][ar * 34 + ac];
            dA[0] = a0.x; dA[1] = a0.y; dA[2] = a0.z; dA[3] = a0.w;
            dA[4] = a1.x; dA[5] = a1.y; dA[6] = a1.z; dA[7] = a1.w;
            const float4 bb[4] = {b0, b1, b2, b3};
#pragma unroll
            for (int p = 0; p < 4; ++p) {
                unsigned int p01 = pack2_bf16_trunc(bb[p].x * dv, bb[p].y * dv);
                unsigned int p23 = pack2_bf16_trunc(bb[p].z * dv, bb[p].w * dv);
                Bsub[nb][(bc + p * 4 + 0) * 34 + br] = (unsigned short)p01;
                Bsub[nb][(bc + p * 4 + 1) * 34 + br] = (unsigned short)(p01 >> 16);
                Bsub[nb][(bc + p * 4 + 2) * 34 + br] = (unsigned short)p23;
                Bsub[nb][(bc + p * 4 + 3) * 34 + br] = (unsigned short)(p23 >> 16);
            }
        }
        buf ^= 1;
    }

    // ---- fused epilogue ----
    const float dtv = dt[0];
    const float c1  = 1.f - dtv;
    const float c2  = 4.f * dtv;      // dt / EPSILON
#pragma unroll
    for (int mi = 0; mi < 2; ++mi) {
#pragma unroll
        for (int v = 0; v < 8; ++v) {
            const int row = bm + wm * 32 + mi * 16 + v + half * 8;
            const float s = c2 / Dd[row];
#pragma unroll
            for (int nj = 0; nj < 4; ++nj) {
                const int col = bn + wn * 64 + nj * 16 + lr;
                const size_t idx = (size_t)row * FF + col;
                out[idx] = c1 * xf[idx] + s * acc[mi][nj][v];
            }
        }
    }
}

// ---------------------------------------------------------------------------
extern "C" void kernel_launch(void* const* d_in, const int* in_sizes, int n_in,
                              void* d_out, int out_size, void* d_ws, size_t ws_size,
                              hipStream_t stream) {
    (void)in_sizes; (void)n_in; (void)out_size; (void)ws_size;
    const float* xf     = (const float*)d_in[0];
    const float* proj_w = (const float*)d_in[1];
    const float* proj_b = (const float*)d_in[2];
    const float* pi_w1  = (const float*)d_in[3];
    const float* pi_b1  = (const float*)d_in[4];
    const float* pi_w2  = (const float*)d_in[5];
    const float* pi_b2  = (const float*)d_in[6];
    const float* dt     = (const float*)d_in[7];
    float* out = (float*)d_out;

    // workspace carve-up (~33.9 MB total)
    char* ws = (char*)d_ws;
    float* z    = (float*)ws;  ws += (size_t)NN * LL * sizeof(float);
    float* sq   = (float*)ws;  ws += (size_t)NN * sizeof(float);
    float* q    = (float*)ws;  ws += (size_t)NN * sizeof(float);
    float* dvec = (float*)ws;  ws += (size_t)NN * sizeof(float);
    float* Dd   = (float*)ws;  ws += (size_t)NN * sizeof(float);
    unsigned short* Kbf = (unsigned short*)ws;   // NN*NN bf16 = 32 MB

    proj_kernel<<<NN, 256, 0, stream>>>(xf, proj_w, proj_b, z, sq);
    kq_kernel  <<<NN, 256, 0, stream>>>(z, sq, Kbf, q);
    pi_kernel  <<<NN, 256, 0, stream>>>(z, pi_w1, pi_b1, pi_w2, pi_b2, q, dvec);
    dsum_kernel<<<NN, 256, 0, stream>>>(Kbf, dvec, Dd);
    gemm_kernel<<<dim3(NN / 128, NN / 128), 256, 0, stream>>>(Kbf, xf, dvec, Dd, dt, out);
}
